// SelectedUnitsHead_65274912964986
// MI455X (gfx1250) — matrix operand order repaired and checked by reading, not compiled, not run on hardware
//
#include <hip/hip_runtime.h>
#include <hip/hip_bf16.h>
#include <math.h>

// ---------------------------------------------------------------------------
// Problem constants (match reference)
// ---------------------------------------------------------------------------
#define BB    1024
#define NN    512
#define EE    256
#define D_IN  1024
#define KK    32
#define HH    32
#define F_DIM 256
#define UT    256
#define TT    64
#define NKEY  513          // N + 1 (end flag)
#define NEGC  1e9f

typedef __attribute__((ext_vector_type(16))) __bf16 v16bf;
typedef __attribute__((ext_vector_type(8)))  float  v8f;

__device__ __forceinline__ __bf16 tobf(float x) { return (__bf16)x; }

// ---------------------------------------------------------------------------
// Generic WMMA GEMM:  Out = act( (A [+A2]) @ W + bias [+ Cadd] )
//   A:[M,Ka] f32 row-major, W:[Ka,Nw] f32 row-major, Out:[M,Nw] f32.
//   Compile-time Ka/Nw => immediate-offset clause loads, no address chains.
//   Each wave owns a 64-row x 32-col macro tile (4 row tiles x 2 col tiles):
//     - A row chunks fetched once as float4 pairs (global_load_b128),
//       feeding BOTH column tiles (for Nw==32 A is read exactly once).
//     - 2 strided B fragments per k-step feed 8 independent
//       v_wmma_f32_16x16x32_bf16 (no D->A/B hazards between them).
//   Requires M%64==0, Ka%32==0, Nw%32==0.
// ---------------------------------------------------------------------------
template <int Ka, int Nw, bool HAS_A2, bool HAS_CADD, bool RELU>
__global__ void __launch_bounds__(256)
wmma_gemm_kernel(const float* __restrict__ A,
                 const float* __restrict__ A2,
                 const float* __restrict__ W,
                 const float* __restrict__ bias,
                 const float* __restrict__ Cadd,
                 float* __restrict__ Out,
                 int total_groups, int groups_n)
{
    const int wave = threadIdx.x >> 5;
    const int lane = threadIdx.x & 31;
    const int g    = blockIdx.x * 8 + wave;
    if (g >= total_groups) return;                   // wave-uniform guard

    const int gn   = g % groups_n;                   // 32-col group
    const int tm0  = (g / groups_n) * 4;             // 4 row tiles per wave
    const int n0   = gn * 32;
    const int m    = lane & 15;                      // A-row / B-col / D-col
    const int kh   = lane >> 4;                      // K-half selector

    const float* __restrict__ arow0  = A + (size_t)(tm0 * 16 + m) * Ka;
    const float* __restrict__ a2row0 = HAS_A2 ? (A2 + (size_t)(tm0 * 16 + m) * Ka) : nullptr;
    // per-lane B base: column (n0+m), K rows kh*16.. ; e*Nw are immediates
    const float* __restrict__ wbase0 = W + (size_t)(kh * 16) * Nw + n0 + m;

    v8f acc[4][2];
#pragma unroll
    for (int j = 0; j < 4; ++j)
#pragma unroll
        for (int c = 0; c < 2; ++c)
            acc[j][c] = (v8f){0.f, 0.f, 0.f, 0.f, 0.f, 0.f, 0.f, 0.f};

    for (int k0 = 0; k0 < Ka; k0 += 32) {
        // ---- two B fragments (cols n0..n0+15 and n0+16..n0+31), clause loads
        const float* __restrict__ wb = wbase0 + (size_t)k0 * Nw;
        v16bf bf0, bf1;
#pragma unroll
        for (int e = 0; e < 16; ++e) {
            bf0[e] = tobf(wb[e * Nw]);
            bf1[e] = tobf(wb[e * Nw + 16]);
        }

        const int ka = k0 + kh * 16;                 // lane's first K chunk
#pragma unroll
        for (int j = 0; j < 4; ++j) {
            // ---- A fragment: two contiguous 8-float chunks -> 4x b128 loads
            const float4* p0 = (const float4*)(arow0 + (size_t)j * 16 * Ka + ka);
            const float4* p1 = (const float4*)(arow0 + (size_t)j * 16 * Ka + ka + 8);
            float4 c0 = p0[0], c1 = p0[1];
            float4 c2 = p1[0], c3 = p1[1];
            if (HAS_A2) {
                const float4* q0 = (const float4*)(a2row0 + (size_t)j * 16 * Ka + ka);
                const float4* q1 = (const float4*)(a2row0 + (size_t)j * 16 * Ka + ka + 8);
                float4 d0 = q0[0], d1 = q0[1], d2 = q1[0], d3 = q1[1];
                c0.x += d0.x; c0.y += d0.y; c0.z += d0.z; c0.w += d0.w;
                c1.x += d1.x; c1.y += d1.y; c1.z += d1.z; c1.w += d1.w;
                c2.x += d2.x; c2.y += d2.y; c2.z += d2.z; c2.w += d2.w;
                c3.x += d3.x; c3.y += d3.y; c3.z += d3.z; c3.w += d3.w;
            }
            v16bf af;
            af[0]  = tobf(c0.x); af[1]  = tobf(c0.y); af[2]  = tobf(c0.z); af[3]  = tobf(c0.w);
            af[4]  = tobf(c1.x); af[5]  = tobf(c1.y); af[6]  = tobf(c1.z); af[7]  = tobf(c1.w);
            af[8]  = tobf(c2.x); af[9]  = tobf(c2.y); af[10] = tobf(c2.z); af[11] = tobf(c2.w);
            af[12] = tobf(c3.x); af[13] = tobf(c3.y); af[14] = tobf(c3.z); af[15] = tobf(c3.w);

            acc[j][0] = __builtin_amdgcn_wmma_f32_16x16x32_bf16(
                            false, af, false, bf0, (short)0, acc[j][0], false, false);
            acc[j][1] = __builtin_amdgcn_wmma_f32_16x16x32_bf16(
                            false, af, false, bf1, (short)0, acc[j][1], false, false);
        }
    }

    // ---- epilogue: bias (+Cadd) (+relu); D layout: M = r + 8*kh, N = lane&15.
    const float bv0 = bias[n0 + m];
    const float bv1 = bias[n0 + 16 + m];
#pragma unroll
    for (int j = 0; j < 4; ++j) {
        const int row0 = (tm0 + j) * 16;
#pragma unroll
        for (int r = 0; r < 8; ++r) {
            const int    orow = row0 + r + 8 * kh;
            const size_t obase = (size_t)orow * Nw + n0 + m;
            float v0 = acc[j][0][r] + bv0;
            float v1 = acc[j][1][r] + bv1;
            if (HAS_CADD) { v0 += Cadd[obase]; v1 += Cadd[obase + 16]; }
            if (RELU)     { v0 = fmaxf(v0, 0.f); v1 = fmaxf(v1, 0.f); }
            Out[obase]      = v0;
            Out[obase + 16] = v1;
        }
    }
}

// ---------------------------------------------------------------------------
// Per-sample LayerNorm-LSTM + pointer-argmax loop.
// One block (256 threads, 8 wave32) per batch element; that sample's
// 513x32 f32 keys live in LDS (row stride 33 to avoid bank conflicts),
// so T=64 steps never touch HBM except the logits stream-out.
// ---------------------------------------------------------------------------
__device__ __forceinline__ float sigf(float x) { return 1.f / (1.f + __expf(-x)); }

__global__ void __launch_bounds__(256)
lstm_pointer_kernel(const float* __restrict__ keyt,        // [B*512, 32]
                    const float* __restrict__ x0_in,       // [B, 32]
                    const float* __restrict__ units_mask,  // [B, 512]
                    const float* __restrict__ Wx,          // [32, 128]
                    const float* __restrict__ Wh,          // [32, 128]
                    const float* __restrict__ b_lstm,      // [128]
                    const float* __restrict__ gxv, const float* __restrict__ bxv,
                    const float* __restrict__ ghv, const float* __restrict__ bhv,
                    float* __restrict__ logits,            // [B, T, 513]
                    float* __restrict__ units_out,         // [B, 513]
                    float* __restrict__ sel_emb)           // [B, 32]
{
    __shared__ float kk[NKEY * 33];        // keys, padded stride
    __shared__ float q[NKEY];
    __shared__ float maskv[NKEY];
    __shared__ float units[NKEY];
    __shared__ float xb[HH], hb[HH], cb[HH];
    __shared__ float xg[4 * HH], hg[4 * HH], gsh[4 * HH];
    __shared__ float red[16];
    __shared__ float smean[2], svar[2];
    __shared__ float rv[8], rav[8];
    __shared__ int   ri[8], rai[8];
    __shared__ int   s_active, s_actrec;

    const int tid  = threadIdx.x;
    const int lane = tid & 31;
    const int wid  = tid >> 5;
    const int b    = blockIdx.x;

    // ---- init: keys -> LDS, end-key row = 0, mask, state
    for (int i = tid; i < NN * KK; i += 256) {
        const int n = i >> 5, k = i & 31;
        kk[n * 33 + k] = keyt[(size_t)b * NN * KK + i];
    }
    if (tid < KK) kk[512 * 33 + tid] = 0.f;
    for (int n = tid; n < NN; n += 256) maskv[n] = units_mask[(size_t)b * NN + n];
    for (int n = tid; n < NKEY; n += 256) units[n] = 0.f;
    if (tid < KK) { xb[tid] = x0_in[(size_t)b * KK + tid]; hb[tid] = 0.f; cb[tid] = 0.f; }
    if (tid == 0) { maskv[512] = 1.f; s_active = 1; }
    __syncthreads();

    for (int t = 0; t < TT; ++t) {
        // ---- gate pre-activations: x@Wx, h@Wh (128 outputs each)
        if (tid < 128) {
            float sx = 0.f, sh = 0.f;
#pragma unroll
            for (int k = 0; k < HH; ++k) {
                sx = fmaf(xb[k], Wx[k * 128 + tid], sx);
                sh = fmaf(hb[k], Wh[k * 128 + tid], sh);
            }
            xg[tid] = sx; hg[tid] = sh;
        }
        __syncthreads();

        // ---- LayerNorm statistics for xg and hg (mean / var over 128)
        for (int which = 0; which < 2; ++which) {
            const float* g = which ? hg : xg;
            float v  = (tid < 128) ? g[tid] : 0.f;
            float s  = v, s2 = v * v;
#pragma unroll
            for (int o = 16; o > 0; o >>= 1) {
                s  += __shfl_down(s,  o, 32);
                s2 += __shfl_down(s2, o, 32);
            }
            if (lane == 0) { red[wid] = s; red[8 + wid] = s2; }
            __syncthreads();
            if (tid == 0) {
                float S = 0.f, Q2 = 0.f;
#pragma unroll
                for (int i = 0; i < 8; ++i) { S += red[i]; Q2 += red[8 + i]; }
                const float mn = S * (1.f / 128.f);
                smean[which] = mn;
                svar[which]  = Q2 * (1.f / 128.f) - mn * mn;
            }
            __syncthreads();
        }

        // ---- gates = LN(x@Wx)*gx+bx + LN(h@Wh)*gh+bh + b_lstm
        if (tid < 128) {
            const float lx = (xg[tid] - smean[0]) * rsqrtf(svar[0] + 1e-5f) * gxv[tid] + bxv[tid];
            const float lh = (hg[tid] - smean[1]) * rsqrtf(svar[1] + 1e-5f) * ghv[tid] + bhv[tid];
            gsh[tid] = lx + lh + b_lstm[tid];
        }
        __syncthreads();
        if (tid < HH) {
            const float ig = gsh[tid], fg = gsh[32 + tid], gg = gsh[64 + tid], og = gsh[96 + tid];
            const float c  = sigf(fg) * cb[tid] + sigf(ig) * tanhf(gg);
            const float h  = sigf(og) * tanhf(c);
            cb[tid] = c; hb[tid] = h; xb[tid] = h;   // next-step x = h
        }
        __syncthreads();

        // ---- score all 513 keys from LDS, track argmax (+alt excluding end)
        float bval = -1e30f; int bidx = 1 << 20;
        float aval = -1e30f; int aidx = 1 << 20;
        for (int n = tid; n < NKEY; n += 256) {
            float d = 0.f;
#pragma unroll
            for (int k = 0; k < KK; ++k) d = fmaf(hb[k], kk[n * 33 + k], d);
            const float qv = d * (1.f / 32.f) - (1.f - maskv[n]) * NEGC;
            q[n] = qv;
            if (qv > bval) { bval = qv; bidx = n; }                 // first-max tie-break
            if (n != 512 && qv > aval) { aval = qv; aidx = n; }
        }
#pragma unroll
        for (int o = 16; o > 0; o >>= 1) {
            float ov = __shfl_down(bval, o, 32); int oi = __shfl_down(bidx, o, 32);
            if (ov > bval || (ov == bval && oi < bidx)) { bval = ov; bidx = oi; }
            ov = __shfl_down(aval, o, 32); oi = __shfl_down(aidx, o, 32);
            if (ov > aval || (ov == aval && oi < aidx)) { aval = ov; aidx = oi; }
        }
        if (lane == 0) { rv[wid] = bval; ri[wid] = bidx; rav[wid] = aval; rai[wid] = aidx; }
        __syncthreads();

        if (tid == 0) {
            float bv2 = rv[0]; int bi2 = ri[0];
            float av2 = rav[0]; int ai2 = rai[0];
#pragma unroll
            for (int i = 1; i < 8; ++i) {
                if (rv[i]  > bv2 || (rv[i]  == bv2 && ri[i]  < bi2)) { bv2 = rv[i];  bi2 = ri[i]; }
                if (rav[i] > av2 || (rav[i] == av2 && rai[i] < ai2)) { av2 = rav[i]; ai2 = rai[i]; }
            }
            const int  sel       = bi2;
            const bool is_end    = (sel == 512);
            const bool first_end = is_end && (t == 0);
            const int  sel_eff   = first_end ? ai2 : sel;
            const int  act       = s_active;
            s_actrec = act;                                   // active BEFORE update
            const bool do_sel = act && (!is_end || first_end);
            if (do_sel) { units[sel_eff] = 1.f; maskv[sel_eff] = 0.f; }
            if (is_end) s_active = 0;
        }
        __syncthreads();

        // ---- stream logits (zeroed after a sample stops)
        const float af_ = (float)s_actrec;
        float* lrow = logits + (size_t)b * TT * NKEY + (size_t)t * NKEY;
        for (int n = tid; n < NKEY; n += 256) lrow[n] = q[n] * af_;
        __syncthreads();
    }

    // ---- outputs: units, sel_emb = sum(units*key)/sum(units)
    for (int n = tid; n < NKEY; n += 256) units_out[(size_t)b * NKEY + n] = units[n];
    if (tid < KK) {
        float s = 0.f, u = 0.f;
        for (int n = 0; n < NKEY; ++n) {
            const float un = units[n];
            u += un;
            s = fmaf(un, kk[n * 33 + tid], s);
        }
        sel_emb[(size_t)b * KK + tid] = s / u;
    }
}

// ---------------------------------------------------------------------------
// Host launcher
// ---------------------------------------------------------------------------
template <int Ka, int Nw, bool HA2, bool HCADD, bool RELU>
static inline void launch_gemm(const float* A, const float* A2, const float* W,
                               const float* bias, const float* Cadd, float* Out,
                               int M, hipStream_t stream) {
    const int groups_n = Nw / 32;
    const int groups   = (M / 64) * groups_n;
    const int blocks   = (groups + 7) / 8;
    hipLaunchKernelGGL((wmma_gemm_kernel<Ka, Nw, HA2, HCADD, RELU>),
                       dim3(blocks), dim3(256), 0, stream,
                       A, A2, W, bias, Cadd, Out, groups, groups_n);
}

extern "C" void kernel_launch(void* const* d_in, const int* in_sizes, int n_in,
                              void* d_out, int out_size, void* d_ws, size_t ws_size,
                              hipStream_t stream) {
    const float* embedding  = (const float*)d_in[0];
    const float* ut_mask    = (const float*)d_in[1];
    const float* units_mask = (const float*)d_in[2];
    const float* entity     = (const float*)d_in[3];
    const float* Wk = (const float*)d_in[4];   const float* bk = (const float*)d_in[5];
    const float* Wf = (const float*)d_in[6];   const float* bf = (const float*)d_in[7];
    const float* W1 = (const float*)d_in[8];   const float* b1 = (const float*)d_in[9];
    const float* W2 = (const float*)d_in[10];  const float* b2 = (const float*)d_in[11];
    const float* We = (const float*)d_in[12];  const float* be = (const float*)d_in[13];
    const float* Wx = (const float*)d_in[14];  const float* Wh = (const float*)d_in[15];
    const float* bl = (const float*)d_in[16];
    const float* gx = (const float*)d_in[17];  const float* bx = (const float*)d_in[18];
    const float* gh = (const float*)d_in[19];  const float* bh = (const float*)d_in[20];

    float* out     = (float*)d_out;
    float* logits  = out;                                          // [B,T,513]
    float* units_o = logits + (size_t)BB * TT * NKEY;              // [B,513]
    float* newemb  = units_o + (size_t)BB * NKEY;                  // [B,1024]

    float* ws    = (float*)d_ws;
    float* keyt  = ws;                                             // [B*512, 32]
    float* x0a   = keyt + (size_t)BB * NN * KK;                    // [B, 256]
    float* func  = x0a  + (size_t)BB * F_DIM;                      // [B, 256]
    float* x0    = func + (size_t)BB * F_DIM;                      // [B, 32]
    float* sel   = x0   + (size_t)BB * KK;                         // [B, 32]

    // keys per entity (end row handled in LDS by the pointer kernel)
    launch_gemm<EE,    KK,    false, false, false>(entity,    nullptr, Wk, bk, nullptr,   keyt,  BB * NN, stream);
    // x0a = relu(embedding @ W1 + b1)
    launch_gemm<D_IN,  F_DIM, false, false, true >(embedding, nullptr, W1, b1, nullptr,   x0a,   BB,      stream);
    // func = relu(unit_type_mask @ Wf + bf)
    launch_gemm<UT,    F_DIM, false, false, true >(ut_mask,   nullptr, Wf, bf, nullptr,   func,  BB,      stream);
    // x0 = relu((x0a + func) @ W2 + b2)
    launch_gemm<F_DIM, KK,    true,  false, true >(x0a,       func,    W2, b2, nullptr,   x0,    BB,      stream);

    // sequential pointer loop: one block per sample, keys resident in LDS
    hipLaunchKernelGGL(lstm_pointer_kernel, dim3(BB), dim3(256), 0, stream,
                       keyt, x0, units_mask, Wx, Wh, bl, gx, bx, gh, bh,
                       logits, units_o, sel);

    // new_embedding = embedding + sel_emb @ We + be
    launch_gemm<KK,    D_IN,  false, true,  false>(sel, nullptr, We, be, embedding, newemb, BB, stream);
}
